// SecondStageDetector_79989470920834
// MI455X (gfx1250) — matrix operand order repaired and checked by reading, not compiled, not run on hardware
//
#include <hip/hip_runtime.h>
#include <hip/hip_bf16.h>
#include <math.h>

// ---------------------------------------------------------------------------
// Types for CDNA5 WMMA
// ---------------------------------------------------------------------------
typedef __bf16 v16bf __attribute__((ext_vector_type(16)));
typedef float  v8f   __attribute__((ext_vector_type(8)));

union Frag16 {
    v16bf bf;
    uint4 q[2];
};

__device__ __forceinline__ unsigned short f2bf(float f) {
    unsigned int u = __float_as_uint(f);
    unsigned int r = u + 0x7FFFu + ((u >> 16) & 1u);   // round-to-nearest-even
    return (unsigned short)(r >> 16);
}

// ---------------------------------------------------------------------------
// Constants from the reference
// ---------------------------------------------------------------------------
#define N_PROP 1024
#define C_FEAT 32
#define ROI    7
#define SR     6
#define SGRID  42            // ROI * SR
#define D_FEAT 1568          // C_FEAT * ROI * ROI
#define HID    2048
#define NMS_K  100
#define IMG_H  360
#define IMG_W  1200
#define BEV_H  700
#define BEV_W  800

// ---------------------------------------------------------------------------
// 1) Box projections (BEV + image)
// ---------------------------------------------------------------------------
__global__ void project_kernel(const float* __restrict__ anchors,
                               const float* __restrict__ P,
                               const int* __restrict__ ishape,
                               float* __restrict__ imgBoxes,
                               float* __restrict__ bevBoxes) {
    int i = blockIdx.x * blockDim.x + threadIdx.x;
    if (i >= N_PROP) return;
    float x  = anchors[i * 6 + 0];
    float y  = anchors[i * 6 + 1];
    float z  = anchors[i * 6 + 2];
    float dx = anchors[i * 6 + 3];
    float dy = anchors[i * 6 + 4];
    float dz = anchors[i * 6 + 5];

    // BEV projection
    bevBoxes[i * 4 + 0] = (x - dx * 0.5f - (-40.0f)) / 0.1f;
    bevBoxes[i * 4 + 1] = (70.0f - (z + dz * 0.5f)) / 0.1f;
    bevBoxes[i * 4 + 2] = (x + dx * 0.5f - (-40.0f)) / 0.1f;
    bevBoxes[i * 4 + 3] = (70.0f - (z - dz * 0.5f)) / 0.1f;

    // Image projection over 8 corners
    float p00 = P[0], p01 = P[1], p02 = P[2],  p03 = P[3];
    float p10 = P[4], p11 = P[5], p12 = P[6],  p13 = P[7];
    float p20 = P[8], p21 = P[9], p22 = P[10], p23 = P[11];

    float umin = 1e30f, vmin = 1e30f, umax = -1e30f, vmax = -1e30f;
    for (int c = 0; c < 8; ++c) {
        float sx = (c & 4) ? 1.0f : -1.0f;
        float sy = (c & 2) ? 1.0f : -1.0f;
        float sz = (c & 1) ? 1.0f : -1.0f;
        float cx = x + sx * dx * 0.5f;
        float cy = y + sy * dy * 0.5f;
        float cz = z + sz * dz * 0.5f;
        float pz = p20 * cx + p21 * cy + p22 * cz + p23;
        float pu = p00 * cx + p01 * cy + p02 * cz + p03;
        float pv = p10 * cx + p11 * cy + p12 * cz + p13;
        pz = fmaxf(pz, 0.1f);
        float u = pu / pz, v = pv / pz;
        umin = fminf(umin, u); umax = fmaxf(umax, u);
        vmin = fminf(vmin, v); vmax = fmaxf(vmax, v);
    }
    float Hf = (float)ishape[0];
    float Wf = (float)ishape[1];
    imgBoxes[i * 4 + 0] = fminf(fmaxf(umin, 0.0f), Wf);
    imgBoxes[i * 4 + 1] = fminf(fmaxf(vmin, 0.0f), Hf);
    imgBoxes[i * 4 + 2] = fminf(fmaxf(umax, 0.0f), Wf);
    imgBoxes[i * 4 + 3] = fminf(fmaxf(vmax, 0.0f), Hf);
}

// ---------------------------------------------------------------------------
// 2) Weight convert + transpose: W[K][N] f32 -> Wt[N][K] bf16
// ---------------------------------------------------------------------------
__global__ void transpose_bf16(const float* __restrict__ W,
                               unsigned short* __restrict__ Wt,
                               int K, int N) {
    long long t = (long long)blockIdx.x * blockDim.x + threadIdx.x;
    long long total = (long long)K * N;
    if (t >= total) return;
    int n = (int)(t / K);
    int k = (int)(t - (long long)n * K);
    Wt[t] = f2bf(W[(long long)k * N + n]);
}

// ---------------------------------------------------------------------------
// 3) Dual ROI-align + fuse -> bf16 activations X[N_PROP][D_FEAT]
// ---------------------------------------------------------------------------
__device__ __forceinline__ float bilerp(const float* __restrict__ f, int W,
                                        int y0, int y1, float wy,
                                        int x0, int x1, float wx) {
    float v00 = f[y0 * W + x0];
    float v01 = f[y0 * W + x1];
    float v10 = f[y1 * W + x0];
    float v11 = f[y1 * W + x1];
    return v00 * (1.0f - wy) * (1.0f - wx) + v01 * (1.0f - wy) * wx +
           v10 * wy * (1.0f - wx)          + v11 * wy * wx;
}

__global__ void roi_fuse_kernel(const float* __restrict__ imgF,
                                const float* __restrict__ bevF,
                                const float* __restrict__ imgBoxes,
                                const float* __restrict__ bevBoxes,
                                const float* __restrict__ img_mask,
                                const float* __restrict__ bev_mask,
                                unsigned short* __restrict__ X) {
    __shared__ int   s_ix0[SGRID], s_ix1[SGRID], s_iy0[SGRID], s_iy1[SGRID];
    __shared__ float s_iwx[SGRID], s_iwy[SGRID];
    __shared__ int   s_bx0[SGRID], s_bx1[SGRID], s_by0[SGRID], s_by1[SGRID];
    __shared__ float s_bwx[SGRID], s_bwy[SGRID];

    int roi = blockIdx.x;
    int tid = threadIdx.x;

    if (tid < 4 * SGRID) {
        int which = tid / SGRID;
        int s     = tid % SGRID;
        float g = ((float)s + 0.5f) / (float)SGRID;
        float lo, hi;
        int dim;
        if (which == 0) { lo = imgBoxes[roi * 4 + 0]; hi = imgBoxes[roi * 4 + 2]; dim = IMG_W; }
        else if (which == 1) { lo = imgBoxes[roi * 4 + 1]; hi = imgBoxes[roi * 4 + 3]; dim = IMG_H; }
        else if (which == 2) { lo = bevBoxes[roi * 4 + 0]; hi = bevBoxes[roi * 4 + 2]; dim = BEV_W; }
        else { lo = bevBoxes[roi * 4 + 1]; hi = bevBoxes[roi * 4 + 3]; dim = BEV_H; }
        lo -= 0.5f; hi -= 0.5f;
        float xs  = lo + g * (hi - lo);
        float x0f = floorf(xs);
        x0f = fminf(fmaxf(x0f, 0.0f), (float)(dim - 1));
        float wx = fminf(fmaxf(xs - x0f, 0.0f), 1.0f);
        int x0i = (int)x0f;
        int x1i = min(x0i + 1, dim - 1);
        if (which == 0)      { s_ix0[s] = x0i; s_ix1[s] = x1i; s_iwx[s] = wx; }
        else if (which == 1) { s_iy0[s] = x0i; s_iy1[s] = x1i; s_iwy[s] = wx; }
        else if (which == 2) { s_bx0[s] = x0i; s_bx1[s] = x1i; s_bwx[s] = wx; }
        else                 { s_by0[s] = x0i; s_by1[s] = x1i; s_bwy[s] = wx; }
    }
    __syncthreads();

    float mi = img_mask[0];
    float mb = bev_mask[0];
    float inv = 1.0f / (36.0f * (mi + mb));

    const int HWi = IMG_H * IMG_W;
    const int HWb = BEV_H * BEV_W;

    for (int j = tid; j < D_FEAT; j += blockDim.x) {
        int c  = j / 49;
        int r  = j - c * 49;
        int py = r / 7;
        int px = r - py * 7;
        const float* fi = imgF + c * HWi;
        const float* fb = bevF + c * HWb;
        float ai = 0.0f, ab = 0.0f;
        for (int sy = 0; sy < SR; ++sy) {
            int gy = py * SR + sy;
            int iy0 = s_iy0[gy], iy1 = s_iy1[gy]; float iwy = s_iwy[gy];
            int by0 = s_by0[gy], by1 = s_by1[gy]; float bwy = s_bwy[gy];
            for (int sx = 0; sx < SR; ++sx) {
                int gx = px * SR + sx;
                ai += bilerp(fi, IMG_W, iy0, iy1, iwy, s_ix0[gx], s_ix1[gx], s_iwx[gx]);
                ab += bilerp(fb, BEV_W, by0, by1, bwy, s_bx0[gx], s_bx1[gx], s_bwx[gx]);
            }
        }
        float fused = (mi * ai + mb * ab) * inv;
        X[(long long)roi * D_FEAT + j] = f2bf(fused);
    }
}

// ---------------------------------------------------------------------------
// 4) bf16 WMMA GEMM with register blocking:
//    each wave computes a 32x64 (MxN) macro-tile = 2 A-frags x 4 B-frags
//    -> 8 WMMAs per 6KB of fragment loads (~21 FLOP/byte).
//    8 waves/block arranged 2(M) x 4(N) -> block tile 64x256.
//    C[M][N] = relu(A[M][K] * Bt[N][K]^T + bias)
// ---------------------------------------------------------------------------
__global__ void gemm_wmma(const unsigned short* __restrict__ A,   // [M][K] bf16
                          const unsigned short* __restrict__ Bt,  // [N][K] bf16
                          const float* __restrict__ bias,         // [N]
                          unsigned short* __restrict__ outBF,     // [M][N] bf16 (if storeBF)
                          float* __restrict__ outF,               // [M][N] f32  (else)
                          int N, int K, int storeBF) {
    int wave = threadIdx.x >> 5;
    int lane = threadIdx.x & 31;
    int lr   = lane & 15;
    int hi   = lane >> 4;
    int wm   = wave >> 2;          // 0..1
    int wn   = wave & 3;           // 0..3

    int tileM = blockIdx.x * 64  + wm * 32;   // 2 x 16 rows
    int tileN = blockIdx.y * 256 + wn * 64;   // 4 x 16 cols

    // A fragment 16x32 in this code's contiguous k order (B uses the same): lane<16 row=lr holds K {0..15},
    // lane>=16 row=lr holds K {16..31}  (two contiguous b128 chunks)
    const unsigned short* aptr0 = A + (long long)(tileM + lr)      * K + 16 * hi;
    const unsigned short* aptr1 = A + (long long)(tileM + 16 + lr) * K + 16 * hi;
    // B fragment 32x16: col = lr, element i -> K = k0 + 16*hi + i (contiguous in Bt row)
    const unsigned short* bptr0 = Bt + (long long)(tileN       + lr) * K + 16 * hi;
    const unsigned short* bptr1 = Bt + (long long)(tileN + 16  + lr) * K + 16 * hi;
    const unsigned short* bptr2 = Bt + (long long)(tileN + 32  + lr) * K + 16 * hi;
    const unsigned short* bptr3 = Bt + (long long)(tileN + 48  + lr) * K + 16 * hi;

    v8f acc[2][4] = {};
    for (int k0 = 0; k0 < K; k0 += 32) {
        Frag16 a0, a1, b0, b1, b2, b3;
        a0.q[0] = *(const uint4*)(aptr0 + k0);
        a0.q[1] = *(const uint4*)(aptr0 + k0 + 8);
        a1.q[0] = *(const uint4*)(aptr1 + k0);
        a1.q[1] = *(const uint4*)(aptr1 + k0 + 8);
        b0.q[0] = *(const uint4*)(bptr0 + k0);
        b0.q[1] = *(const uint4*)(bptr0 + k0 + 8);
        b1.q[0] = *(const uint4*)(bptr1 + k0);
        b1.q[1] = *(const uint4*)(bptr1 + k0 + 8);
        b2.q[0] = *(const uint4*)(bptr2 + k0);
        b2.q[1] = *(const uint4*)(bptr2 + k0 + 8);
        b3.q[0] = *(const uint4*)(bptr3 + k0);
        b3.q[1] = *(const uint4*)(bptr3 + k0 + 8);

        acc[0][0] = __builtin_amdgcn_wmma_f32_16x16x32_bf16(false, a0.bf, false, b0.bf, (short)0, acc[0][0], false, false);
        acc[0][1] = __builtin_amdgcn_wmma_f32_16x16x32_bf16(false, a0.bf, false, b1.bf, (short)0, acc[0][1], false, false);
        acc[0][2] = __builtin_amdgcn_wmma_f32_16x16x32_bf16(false, a0.bf, false, b2.bf, (short)0, acc[0][2], false, false);
        acc[0][3] = __builtin_amdgcn_wmma_f32_16x16x32_bf16(false, a0.bf, false, b3.bf, (short)0, acc[0][3], false, false);
        acc[1][0] = __builtin_amdgcn_wmma_f32_16x16x32_bf16(false, a1.bf, false, b0.bf, (short)0, acc[1][0], false, false);
        acc[1][1] = __builtin_amdgcn_wmma_f32_16x16x32_bf16(false, a1.bf, false, b1.bf, (short)0, acc[1][1], false, false);
        acc[1][2] = __builtin_amdgcn_wmma_f32_16x16x32_bf16(false, a1.bf, false, b2.bf, (short)0, acc[1][2], false, false);
        acc[1][3] = __builtin_amdgcn_wmma_f32_16x16x32_bf16(false, a1.bf, false, b3.bf, (short)0, acc[1][3], false, false);
    }

    #pragma unroll
    for (int i = 0; i < 2; ++i) {
        #pragma unroll
        for (int j = 0; j < 4; ++j) {
            int col = tileN + j * 16 + lr;
            float bv = bias[col];
            #pragma unroll
            for (int v = 0; v < 8; ++v) {
                int row = tileM + i * 16 + v + 8 * hi;
                float x = fmaxf(acc[i][j][v] + bv, 0.0f);   // relu for both GEMMs
                if (storeBF) outBF[(long long)row * N + col] = f2bf(x);
                else         outF [(long long)row * N + col] = x;
            }
        }
    }
}

// ---------------------------------------------------------------------------
// 5) Heads: obj(2) / off(10) / ang(2) + softmax, atan2, box decode
// ---------------------------------------------------------------------------
__global__ void heads_kernel(const float* __restrict__ h2,
                             const float* __restrict__ Wc, const float* __restrict__ bc,
                             const float* __restrict__ Wo, const float* __restrict__ bo,
                             const float* __restrict__ Wa, const float* __restrict__ ba,
                             const float* __restrict__ anchors,
                             float* __restrict__ objsoft,
                             float* __restrict__ off10,
                             float* __restrict__ predA,
                             float* __restrict__ predBev,
                             float* __restrict__ scores,
                             float* __restrict__ orient) {
    int row = blockIdx.x;
    int t   = threadIdx.x;

    float acc[14];
    #pragma unroll
    for (int j = 0; j < 14; ++j) acc[j] = 0.0f;

    for (int k = t; k < HID; k += 256) {
        float hv = h2[(long long)row * HID + k];
        acc[0]  += hv * Wc[k * 2 + 0];
        acc[1]  += hv * Wc[k * 2 + 1];
        #pragma unroll
        for (int j = 0; j < 10; ++j) acc[2 + j] += hv * Wo[k * 10 + j];
        acc[12] += hv * Wa[k * 2 + 0];
        acc[13] += hv * Wa[k * 2 + 1];
    }
    // wave32 reduction
    #pragma unroll
    for (int j = 0; j < 14; ++j)
        for (int m = 16; m > 0; m >>= 1)
            acc[j] += __shfl_xor(acc[j], m, 32);

    __shared__ float partial[8][14];
    int wave = t >> 5, lane = t & 31;
    if (lane == 0)
        for (int j = 0; j < 14; ++j) partial[wave][j] = acc[j];
    __syncthreads();

    if (t == 0) {
        float r[14];
        for (int j = 0; j < 14; ++j) {
            float s = 0.0f;
            for (int w = 0; w < 8; ++w) s += partial[w][j];
            r[j] = s;
        }
        float o0 = r[0] + bc[0], o1 = r[1] + bc[1];
        float mx = fmaxf(o0, o1);
        float e0 = expf(o0 - mx), e1 = expf(o1 - mx);
        float se = e0 + e1;
        objsoft[row * 2 + 0] = e0 / se;
        objsoft[row * 2 + 1] = e1 / se;
        scores[row] = o1;                       // obj[:,1:].max == obj[:,1]

        float off[10];
        for (int j = 0; j < 10; ++j) {
            off[j] = r[2 + j] + bo[j];
            off10[row * 10 + j] = off[j];
        }
        float a0 = r[12] + ba[0], a1 = r[13] + ba[1];
        orient[row] = atan2f(a1, a0);

        float pa[6];
        for (int j = 0; j < 6; ++j) {
            pa[j] = anchors[row * 6 + j] + off[j];
            predA[row * 6 + j] = pa[j];
        }
        float x = pa[0], z = pa[2], dx = pa[3], dz = pa[5];
        predBev[row * 4 + 0] = (x - dx * 0.5f + 40.0f) / 0.1f;
        predBev[row * 4 + 1] = (70.0f - (z + dz * 0.5f)) / 0.1f;
        predBev[row * 4 + 2] = (x + dx * 0.5f + 40.0f) / 0.1f;
        predBev[row * 4 + 3] = (70.0f - (z - dz * 0.5f)) / 0.1f;
    }
}

// ---------------------------------------------------------------------------
// 6) NMS: one 1024-thread block, 100 serial argmax+suppress iterations
// ---------------------------------------------------------------------------
__global__ void nms_kernel(const float* __restrict__ boxes,
                           const float* __restrict__ scores_in,
                           int* __restrict__ idx_out) {
    __shared__ float s[N_PROP];
    __shared__ float rv[N_PROP];
    __shared__ int   ri[N_PROP];

    int t = threadIdx.x;
    s[t] = scores_in[t];
    float bx1 = boxes[t * 4 + 0], by1 = boxes[t * 4 + 1];
    float bx2 = boxes[t * 4 + 2], by2 = boxes[t * 4 + 3];
    float area = (bx2 - bx1) * (by2 - by1);
    __syncthreads();

    for (int k = 0; k < NMS_K; ++k) {
        rv[t] = s[t];
        ri[t] = t;
        __syncthreads();
        for (int off = N_PROP / 2; off > 0; off >>= 1) {
            if (t < off) {
                float v2 = rv[t + off];
                int   i2 = ri[t + off];
                if (v2 > rv[t] || (v2 == rv[t] && i2 < ri[t])) { rv[t] = v2; ri[t] = i2; }
            }
            __syncthreads();
        }
        int best = ri[0];
        if (t == 0) idx_out[k] = best;

        float Bx1 = boxes[best * 4 + 0], By1 = boxes[best * 4 + 1];
        float Bx2 = boxes[best * 4 + 2], By2 = boxes[best * 4 + 3];
        float Barea = (Bx2 - Bx1) * (By2 - By1);
        float xx1 = fmaxf(bx1, Bx1), yy1 = fmaxf(by1, By1);
        float xx2 = fminf(bx2, Bx2), yy2 = fminf(by2, By2);
        float inter = fmaxf(xx2 - xx1, 0.0f) * fmaxf(yy2 - yy1, 0.0f);
        float iou = inter / (area + Barea - inter + 1e-6f);
        __syncthreads();
        if (iou > 0.01f) s[t] = -INFINITY;
        if (t == best)   s[t] = -INFINITY;
        __syncthreads();
    }
}

// ---------------------------------------------------------------------------
// 7) Gather selected detections into d_out
//    layout: objsoft[100*2] | predA[100*6] | off[100*10] | pbox[100*7] | orient[100]
// ---------------------------------------------------------------------------
__global__ void gather_kernel(const int* __restrict__ idx,
                              const float* __restrict__ objsoft,
                              const float* __restrict__ predA,
                              const float* __restrict__ off10,
                              const float* __restrict__ orient,
                              float* __restrict__ out) {
    int k = threadIdx.x;
    if (k >= NMS_K) return;
    int i = idx[k];
    out[k * 2 + 0] = objsoft[i * 2 + 0];
    out[k * 2 + 1] = objsoft[i * 2 + 1];
    for (int j = 0; j < 6; ++j)  out[200  + k * 6  + j] = predA[i * 6 + j];
    for (int j = 0; j < 10; ++j) out[800  + k * 10 + j] = off10[i * 10 + j];
    for (int j = 0; j < 6; ++j)  out[1800 + k * 7  + j] = predA[i * 6 + j];
    out[1800 + k * 7 + 6] = 0.0f;
    out[2500 + k] = orient[i];
}

// ---------------------------------------------------------------------------
// Launcher
// ---------------------------------------------------------------------------
extern "C" void kernel_launch(void* const* d_in, const int* in_sizes, int n_in,
                              void* d_out, int out_size, void* d_ws, size_t ws_size,
                              hipStream_t stream) {
    (void)in_sizes; (void)n_in; (void)out_size; (void)ws_size;

    const float* imgF     = (const float*)d_in[0];
    const float* bevF     = (const float*)d_in[1];
    const float* anchors  = (const float*)d_in[2];
    const float* calib    = (const float*)d_in[3];
    /* ground_plane d_in[4] multiplied by 0 in reference -> unused */
    const float* img_mask = (const float*)d_in[5];
    const float* bev_mask = (const float*)d_in[6];
    const float* W1       = (const float*)d_in[7];
    const float* b1       = (const float*)d_in[8];
    const float* W2       = (const float*)d_in[9];
    const float* b2       = (const float*)d_in[10];
    const float* Wc       = (const float*)d_in[11];
    const float* bc       = (const float*)d_in[12];
    const float* Wo       = (const float*)d_in[13];
    const float* bo       = (const float*)d_in[14];
    const float* Wa       = (const float*)d_in[15];
    const float* ba       = (const float*)d_in[16];
    const int*   ishape   = (const int*)d_in[17];
    float* out = (float*)d_out;

    // workspace carve-out (256B aligned chunks)
    char* p = (char*)d_ws;
    auto alloc = [&](size_t bytes) -> void* {
        void* r = (void*)p;
        p += (bytes + 255) & ~(size_t)255;
        return r;
    };
    unsigned short* X    = (unsigned short*)alloc((size_t)N_PROP * D_FEAT * 2);
    unsigned short* W1t  = (unsigned short*)alloc((size_t)HID * D_FEAT * 2);
    unsigned short* W2t  = (unsigned short*)alloc((size_t)HID * HID * 2);
    unsigned short* h1   = (unsigned short*)alloc((size_t)N_PROP * HID * 2);
    float* h2      = (float*)alloc((size_t)N_PROP * HID * 4);
    float* imgB    = (float*)alloc((size_t)N_PROP * 4 * 4);
    float* bevB    = (float*)alloc((size_t)N_PROP * 4 * 4);
    float* objsoft = (float*)alloc((size_t)N_PROP * 2 * 4);
    float* off10   = (float*)alloc((size_t)N_PROP * 10 * 4);
    float* predA   = (float*)alloc((size_t)N_PROP * 6 * 4);
    float* predBev = (float*)alloc((size_t)N_PROP * 4 * 4);
    float* scores  = (float*)alloc((size_t)N_PROP * 4);
    float* orient  = (float*)alloc((size_t)N_PROP * 4);
    int*   nmsIdx  = (int*)alloc(256 * 4);

    // 1) projections
    project_kernel<<<dim3((N_PROP + 255) / 256), dim3(256), 0, stream>>>(
        anchors, calib, ishape, imgB, bevB);

    // 2) weight convert+transpose to bf16
    {
        long long t1 = (long long)HID * D_FEAT;
        transpose_bf16<<<dim3((unsigned)((t1 + 255) / 256)), dim3(256), 0, stream>>>(
            W1, W1t, D_FEAT, HID);
        long long t2 = (long long)HID * HID;
        transpose_bf16<<<dim3((unsigned)((t2 + 255) / 256)), dim3(256), 0, stream>>>(
            W2, W2t, HID, HID);
    }

    // 3) dual ROI-align + fuse -> bf16 activations
    roi_fuse_kernel<<<dim3(N_PROP), dim3(256), 0, stream>>>(
        imgF, bevF, imgB, bevB, img_mask, bev_mask, X);

    // 4) GEMM1: h1 = relu(X @ W1 + b1), bf16 out   (block tile 64x256)
    gemm_wmma<<<dim3(N_PROP / 64, HID / 256), dim3(256), 0, stream>>>(
        X, W1t, b1, h1, nullptr, HID, D_FEAT, 1);

    // 5) GEMM2: h2 = relu(h1 @ W2 + b2), f32 out
    gemm_wmma<<<dim3(N_PROP / 64, HID / 256), dim3(256), 0, stream>>>(
        h1, W2t, b2, nullptr, h2, HID, HID, 0);

    // 6) heads + decode
    heads_kernel<<<dim3(N_PROP), dim3(256), 0, stream>>>(
        h2, Wc, bc, Wo, bo, Wa, ba, anchors,
        objsoft, off10, predA, predBev, scores, orient);

    // 7) NMS
    nms_kernel<<<dim3(1), dim3(N_PROP), 0, stream>>>(predBev, scores, nmsIdx);

    // 8) gather outputs
    gather_kernel<<<dim3(1), dim3(128), 0, stream>>>(
        nmsIdx, objsoft, predA, off10, orient, out);
}
